// TSP_net_12532714570262
// MI455X (gfx1250) — matrix operand order repaired and checked by reading, compiled, not run on hardware
//
#include <hip/hip_runtime.h>
#include <hip/hip_bf16.h>
#include <math.h>

// ---------------------------------------------------------------------------
// TSP pointer-network autoregressive decode for MI455X (gfx1250).
// One persistent workgroup per batch element (64 blocks x 256 threads = 8 waves).
// Activations live in LDS; weights pre-packed into WMMA B-fragment layout in the
// workspace (f16) so each fragment is one aligned 32B global load per lane.
// All tall GEMMs use v_wmma_f32_16x16x32_f16 with f32 accumulation.
// ---------------------------------------------------------------------------

typedef __attribute__((ext_vector_type(16))) _Float16 v16h;
typedef __attribute__((ext_vector_type(8)))  _Float16 v8h;
typedef __attribute__((ext_vector_type(8)))  float    v8f;

#define TD    128   // model dim
#define TFF   512   // feed-forward dim
#define NH    8     // heads
#define HDIM  16    // head dim
#define NLE_  2     // encoder layers
#define KA    15    // action_k
#define KS_   35    // state_k
#define NNODE 50
#define NB    64
#define SPAD  48    // padded token rows (max real tokens = 37)
#define BIGF  1.0e9f

// force global (addrspace 1) loads for weight/bias reads
__device__ __forceinline__ float gloadf(const float* p) {
  return *(const __attribute__((address_space(1))) float*)(unsigned long long)p;
}
__device__ __forceinline__ v16h gload16(const _Float16* p) {
  return *(const __attribute__((address_space(1))) v16h*)(unsigned long long)p;
}

// ---- prologue: pack f32 KxN row-major weights into f16 WMMA B-fragment tiles.
// dst element ((kt*Nt + nt)*32 + lane)*16 + e  <=  src[kt*32 + (lane&16) + e][nt*16 + (lane&15)]
// (contiguous K order, same as gemm_f16's A loads: lanes 0-15 hold K 0..15, lanes 16-31 hold K 16..31,
//  column N = lane&15, halves e = K within half-group)
__global__ void pack_bfrag(const float* __restrict__ src, _Float16* __restrict__ dst,
                           int K, int N) {
  int o = blockIdx.x * blockDim.x + threadIdx.x;
  if (o >= K * N) return;
  const int Nt   = N >> 4;
  const int e    = o & 15;
  const int lane = (o >> 4) & 31;
  const int t    = o >> 9;            // kt*Nt + nt
  const int kt   = t / Nt, nt = t % Nt;
  const int k = kt * 32 + (lane & 16) + e;
  const int n = nt * 16 + (lane & 15);
  dst[o] = (_Float16)src[(size_t)k * N + n];
}

// ---- WMMA GEMM: C(MT*16 x Nt*16) = act( A @ Wpacked + bias ) ----
// A: LDS f16 row-major (lda), rows padded to MT*16.  Wf: packed B fragments.
// NtTot = packed matrix width in 16-col tiles; ktOff/ntOff = tile offsets into pack.
// flags: bit0 = relu, bit1 = accumulate into C32.
template <int MT>
__device__ __forceinline__ void gemm_f16(
    const _Float16* A, int lda,
    const _Float16* __restrict__ Wf, int NtTot, int ktOff, int ntOff,
    const float* __restrict__ bias,
    int K, int Nt,
    float* C32, _Float16* C16, int ldc, int flags)
{
  const int lane = threadIdx.x & 31;
  const int wave = threadIdx.x >> 5;
  const int lo   = lane & 15;
  const int hiA  = (lane & 16) >> 1;   // 0 or 8: C row offset; A K-offset is 2*hiA
  for (int nt = wave; nt < Nt; nt += 8) {
    const int n = nt * 16 + lo;
    v8f acc[MT];
#pragma unroll
    for (int mt = 0; mt < MT; ++mt)
#pragma unroll
      for (int r = 0; r < 8; ++r) acc[mt][r] = 0.0f;

    for (int kb = 0; kb < K; kb += 32) {
      // one aligned 32B global load per lane = whole B fragment (L2 resident)
      const v16h b = gload16(Wf + (((size_t)(ktOff + (kb >> 5)) * NtTot + ntOff + nt) * 32
                                   + lane) * 16);
#pragma unroll
      for (int mt = 0; mt < MT; ++mt) {
        const int m = mt * 16 + lo;
        const v8h a0 = *(const v8h*)(A + m * lda + kb + 2 * hiA);
        const v8h a1 = *(const v8h*)(A + m * lda + kb + 2 * hiA + 8);
        const v16h a = __builtin_shufflevector(a0, a1, 0, 1, 2, 3, 4, 5, 6, 7,
                                               8, 9, 10, 11, 12, 13, 14, 15);
        acc[mt] = __builtin_amdgcn_wmma_f32_16x16x32_f16(false, a, false, b, (short)0,
                                                         acc[mt], false, false);
      }
    }
    const float bv = bias ? gloadf(bias + n) : 0.0f;
#pragma unroll
    for (int mt = 0; mt < MT; ++mt) {
#pragma unroll
      for (int r = 0; r < 8; ++r) {
        const int mr = mt * 16 + r + hiA;
        float v = acc[mt][r] + bv;
        if (flags & 1) v = v > 0.0f ? v : 0.0f;
        if (C32) { if (flags & 2) C32[mr * ldc + n] += v; else C32[mr * ldc + n] = v; }
        if (C16) C16[mr * ldc + n] = (_Float16)v;
      }
    }
  }
}

// ---- LayerNorm over rows [0,S) of h32 (row width TD) ----
__device__ __forceinline__ void layer_norm_rows(int S, float* h32, float* mus, float* rss,
                                                const float* __restrict__ g,
                                                const float* __restrict__ bta)
{
  const int tid = threadIdx.x;
  if (tid < S) {
    float mu = 0.0f;
    for (int d = 0; d < TD; ++d) mu += h32[tid * TD + d];
    mu *= (1.0f / TD);
    float var = 0.0f;
    for (int d = 0; d < TD; ++d) { float t = h32[tid * TD + d] - mu; var += t * t; }
    var *= (1.0f / TD);
    mus[tid] = mu;
    rss[tid] = rsqrtf(var + 1e-5f);
  }
  __syncthreads();
  for (int i = tid; i < S * TD; i += 256) {
    int r = i / TD, d = i % TD;
    h32[i] = gloadf(g + d) * (h32[i] - mus[r]) * rss[r] + gloadf(bta + d);
  }
  __syncthreads();
}

// ---- 2-layer transformer encoder over S tokens (h32 rows 0..S-1 populated) ----
__device__ void encode_seq(
    int S,
    float* h32, _Float16* a16, _Float16* q16, _Float16* k16, _Float16* v16,
    float* sc, float* mus, float* rss,
    const _Float16* __restrict__ attW, const float* __restrict__ attB,
    const _Float16* __restrict__ ffW1, const float* __restrict__ ffB1,
    const _Float16* __restrict__ ffW2, const float* __restrict__ ffB2,
    const float* __restrict__ ln)
{
  const int tid = threadIdx.x;
  for (int l = 0; l < NLE_; ++l) {
    // stage residual stream as f16 A-matrix
    for (int i = tid; i < SPAD * TD; i += 256) a16[i] = (_Float16)h32[i];
    __syncthreads();
    // packed attW holds 8 matrices of 128x128 => 4 K-tiles each, 8 N-tiles wide
    gemm_f16<3>(a16, TD, attW, 8, (l*4+0)*4, 0, attB + (l*4+0)*TD, TD, 8,
                nullptr, q16, TD, 0);
    gemm_f16<3>(a16, TD, attW, 8, (l*4+1)*4, 0, attB + (l*4+1)*TD, TD, 8,
                nullptr, k16, TD, 0);
    gemm_f16<3>(a16, TD, attW, 8, (l*4+2)*4, 0, attB + (l*4+2)*TD, TD, 8,
                nullptr, v16, TD, 0);
    __syncthreads();
    // multi-head self-attention, head-sequential; output overwrites a16
    for (int hh = 0; hh < NH; ++hh) {
      for (int idx = tid; idx < S * S; idx += 256) {
        const int i = idx / S, j = idx % S;
        float s = 0.0f;
#pragma unroll
        for (int d = 0; d < HDIM; ++d)
          s += (float)q16[i*TD + hh*HDIM + d] * (float)k16[j*TD + hh*HDIM + d];
        sc[i * 40 + j] = s * 0.25f;   // 1/sqrt(16)
      }
      __syncthreads();
      if (tid < S) {
        float* row = sc + tid * 40;
        float mx = -3.0e38f;
        for (int j = 0; j < S; ++j) mx = fmaxf(mx, row[j]);
        float sum = 0.0f;
        for (int j = 0; j < S; ++j) { float e = expf(row[j] - mx); row[j] = e; sum += e; }
        const float inv = 1.0f / sum;
#pragma unroll
        for (int d = 0; d < HDIM; ++d) {
          float o = 0.0f;
          for (int j = 0; j < S; ++j) o += row[j] * (float)v16[j*TD + hh*HDIM + d];
          a16[tid*TD + hh*HDIM + d] = (_Float16)(o * inv);
        }
      }
      __syncthreads();
    }
    // output projection, accumulated into residual h32
    gemm_f16<3>(a16, TD, attW, 8, (l*4+3)*4, 0, attB + (l*4+3)*TD, TD, 8,
                h32, nullptr, TD, 2);
    __syncthreads();
    layer_norm_rows(S, h32, mus, rss, ln + ((l*2+0)*2+0)*TD, ln + ((l*2+0)*2+1)*TD);
    // feed-forward, chunked over FF (4 chunks of 128), f32 accumulation into h32
    for (int i = tid; i < SPAD * TD; i += 256) a16[i] = (_Float16)h32[i];
    __syncthreads();
    for (int cc = 0; cc < TFF / TD; ++cc) {
      // ffW1 packed as (256 x 512): NtTot=32, layer l at ktOff=l*4, chunk at ntOff=cc*8
      gemm_f16<3>(a16, TD, ffW1, 32, l*4, cc*8, ffB1 + l*TFF + cc*TD, TD, 8,
                  nullptr, k16, TD, 1 /*relu*/);
      __syncthreads();
      // ffW2 packed as (1024 x 128): NtTot=8, rows l*512 + cc*128 => ktOff=l*16+cc*4
      gemm_f16<3>(k16, TD, ffW2, 8, l*16 + cc*4, 0,
                  (cc == 0) ? (ffB2 + l*TD) : nullptr, TD, 8,
                  h32, nullptr, TD, 2 /*accumulate*/);
      __syncthreads();
    }
    layer_norm_rows(S, h32, mus, rss, ln + ((l*2+1)*2+0)*TD, ln + ((l*2+1)*2+1)*TD);
  }
}

// ---- main persistent kernel: one block = one batch element, full 49-step scan ----
__global__ __launch_bounds__(256, 1)
void tsp_solver_kernel(
    const float* __restrict__ x, const int* __restrict__ start_idx,
    const float* __restrict__ aEmbW, const float* __restrict__ aEmbB,
    const _Float16* __restrict__ aAttW, const float* __restrict__ aAttB,
    const _Float16* __restrict__ aFF1, const float* __restrict__ aFB1,
    const _Float16* __restrict__ aFF2, const float* __restrict__ aFB2,
    const float* __restrict__ aLN,
    const float* __restrict__ sEmbW, const float* __restrict__ sEmbB,
    const _Float16* __restrict__ sAttW, const float* __restrict__ sAttB,
    const _Float16* __restrict__ sFF1, const float* __restrict__ sFB1,
    const _Float16* __restrict__ sFF2, const float* __restrict__ sFB2,
    const float* __restrict__ sLN,
    const _Float16* __restrict__ WK16, const float* __restrict__ bK,
    const _Float16* __restrict__ WV16, const float* __restrict__ bV,
    const float* __restrict__ WqMlp, const float* __restrict__ bqMlp,
    const float* __restrict__ dWq, const float* __restrict__ dbq,
    const float* __restrict__ dWo, const float* __restrict__ dbo,
    float* __restrict__ out)
{
  __shared__ float    h32[SPAD * TD];
  __shared__ __align__(32) _Float16 a16[SPAD * TD];
  __shared__ __align__(32) _Float16 q16[SPAD * TD];
  __shared__ __align__(32) _Float16 k16[SPAD * TD];
  __shared__ __align__(32) _Float16 v16[SPAD * TD];
  __shared__ float    sc[SPAD * 40];
  __shared__ __align__(32) _Float16 eo16[16 * 256];  // concat(emb_a, emb_s), padded to 16 rows
  __shared__ __align__(32) _Float16 Kc16[16 * 256];  // per-candidate K, NLD*D
  __shared__ __align__(32) _Float16 Vc16[16 * 256];
  __shared__ float    embq[3 * TD];
  __shared__ float    hvec[TD], qvec[TD], ovec[TD];
  __shared__ float    xy[NNODE * 2];
  __shared__ float    d2s[64];
  __shared__ int      mask_s[64];
  __shared__ int      knn_s[KS_];
  __shared__ int      valid_s[KA];
  __shared__ float    logit_s[KA];
  __shared__ float    mus[SPAD], rss[SPAD];
  __shared__ int      s_last;
  __shared__ float    s_logp;

  const int b   = blockIdx.x;
  const int tid = threadIdx.x;
  const int startn = start_idx[b];

  for (int i = tid; i < NNODE * 2; i += 256) xy[i] = x[(size_t)b * NNODE * 2 + i];
  for (int i = tid; i < NNODE; i += 256) mask_s[i] = 1;
  __syncthreads();
  if (tid == 0) {
    mask_s[startn] = 0;
    s_last = startn;
    s_logp = 0.0f;
    out[(size_t)b * NNODE] = (float)startn;
  }
  __syncthreads();

  for (int step = 0; step < NNODE - 1; ++step) {
    const int lastn = s_last;
    const float lx = xy[2 * lastn], ly = xy[2 * lastn + 1];

    // squared distances with visited-mask
    if (tid < NNODE) {
      float dx = xy[2 * tid] - lx, dy = xy[2 * tid + 1] - ly;
      float dd = dx * dx + dy * dy;
      d2s[tid] = mask_s[tid] ? dd : BIGF;
    }
    __syncthreads();
    // top-35 nearest (serial selection; ties -> lowest index, matches lax.top_k)
    if (tid == 0) {
      for (int t = 0; t < KS_; ++t) {
        float best = 3.0e38f; int bi = 0;
        for (int j = 0; j < NNODE; ++j)
          if (d2s[j] < best) { best = d2s[j]; bi = j; }
        knn_s[t] = bi;
        d2s[bi] = 3.9e38f;
      }
    }
    __syncthreads();
    if (tid < KA) valid_s[tid] = mask_s[knn_s[tid]];

    // ---- action encoder: tokens = 15 candidates + last ----
    for (int i = tid; i < (KA + 1) * TD; i += 256) {
      int t = i / TD, d = i % TD;
      int node = (t < KA) ? knn_s[t] : lastn;
      h32[i] = xy[2*node] * gloadf(aEmbW + d) + xy[2*node+1] * gloadf(aEmbW + TD + d)
             + gloadf(aEmbB + d);
    }
    __syncthreads();
    encode_seq(KA + 1, h32, a16, q16, k16, v16, sc, mus, rss,
               aAttW, aAttB, aFF1, aFB1, aFF2, aFB2, aLN);
    if (tid < TD) embq[tid] = h32[KA * TD + tid];
    for (int i = tid; i < KA * TD; i += 256)
      eo16[(i / TD) * 256 + (i % TD)] = (_Float16)h32[i];
    __syncthreads();

    // ---- state encoder: tokens = 35 candidates + last + first ----
    for (int i = tid; i < (KS_ + 2) * TD; i += 256) {
      int t = i / TD, d = i % TD;
      int node = (t < KS_) ? knn_s[t] : ((t == KS_) ? lastn : startn);
      h32[i] = xy[2*node] * gloadf(sEmbW + d) + xy[2*node+1] * gloadf(sEmbW + TD + d)
             + gloadf(sEmbB + d);
    }
    __syncthreads();
    encode_seq(KS_ + 2, h32, a16, q16, k16, v16, sc, mus, rss,
               sAttW, sAttB, sFF1, sFB1, sFF2, sFB2, sLN);
    if (tid < TD) {
      embq[TD + tid]     = h32[KS_ * TD + tid];
      embq[2 * TD + tid] = h32[(KS_ + 1) * TD + tid];
    }
    for (int i = tid; i < KA * TD; i += 256)
      eo16[(i / TD) * 256 + TD + (i % TD)] = (_Float16)h32[i];
    eo16[KA * 256 + tid] = (_Float16)0.0f;  // zero padded candidate row 15
    __syncthreads();

    // h = emb_q @ Wq_mlp + b (1x384 @ 384x128, VALU)
    if (tid < TD) {
      float acc = gloadf(bqMlp + tid);
      for (int k = 0; k < 3 * TD; ++k) acc += embq[k] * gloadf(WqMlp + k * TD + tid);
      hvec[tid] = acc;
    }
    // K,V projections: (16x256)@(256x256) via WMMA; packed NtTot=16
    gemm_f16<1>(eo16, 256, WK16, 16, 0, 0, bK, 256, 16, nullptr, Kc16, 256, 0);
    gemm_f16<1>(eo16, 256, WV16, 16, 0, 0, bV, 256, 16, nullptr, Vc16, 256, 0);
    __syncthreads();

    // ---- decoder layer 0: cross-attention over 15 candidates ----
    if (tid < TD) {
      float acc = gloadf(dbq + tid);
      for (int k = 0; k < TD; ++k) acc += hvec[k] * gloadf(dWq + k * TD + tid);
      qvec[tid] = acc;
    }
    __syncthreads();
    if (tid < NH) {
      float scl[KA];
      float mx = -3.0e38f;
      for (int kk = 0; kk < KA; ++kk) {
        float s = 0.0f;
        for (int d = 0; d < HDIM; ++d)
          s += qvec[tid*HDIM + d] * (float)Kc16[kk*256 + tid*HDIM + d];
        s *= 0.25f;
        if (!valid_s[kk]) s = -BIGF;
        scl[kk] = s;
        mx = fmaxf(mx, s);
      }
      float sum = 0.0f;
      for (int kk = 0; kk < KA; ++kk) { scl[kk] = expf(scl[kk] - mx); sum += scl[kk]; }
      const float inv = 1.0f / sum;
      for (int d = 0; d < HDIM; ++d) {
        float o = 0.0f;
        for (int kk = 0; kk < KA; ++kk)
          o += scl[kk] * (float)Vc16[kk*256 + tid*HDIM + d];
        ovec[tid*HDIM + d] = o * inv;
      }
    }
    __syncthreads();
    if (tid < TD) {
      float acc = gloadf(dbo + tid);
      for (int k = 0; k < TD; ++k) acc += ovec[k] * gloadf(dWo + k * TD + tid);
      hvec[tid] += acc;
    }
    __syncthreads();
    // ---- pointer head ----
    if (tid < TD) {
      float acc = gloadf(dbq + TD + tid);
      for (int k = 0; k < TD; ++k) acc += hvec[k] * gloadf(dWq + (TD + k) * TD + tid);
      qvec[tid] = acc;
    }
    __syncthreads();
    if (tid < KA) {
      float s = 0.0f;
      for (int d = 0; d < TD; ++d) s += qvec[d] * (float)Kc16[tid*256 + TD + d];
      s *= 0.08838834764831845f;       // 1/sqrt(128)
      float lg = 10.0f * tanhf(s);
      logit_s[tid] = valid_s[tid] ? lg : -BIGF;
    }
    __syncthreads();
    if (tid == 0) {
      float mx = -3.0e38f; int bi = 0;
      for (int kk = 0; kk < KA; ++kk)
        if (logit_s[kk] > mx) { mx = logit_s[kk]; bi = kk; }
      float sum = 0.0f;
      for (int kk = 0; kk < KA; ++kk) sum += expf(logit_s[kk] - mx);
      s_logp += (logit_s[bi] - mx) - logf(sum);
      const int nxt = knn_s[bi];
      mask_s[nxt] = 0;
      s_last = nxt;
      out[(size_t)b * NNODE + step + 1] = (float)nxt;
    }
    __syncthreads();
  }
  if (tid == 0) out[(size_t)NB * NNODE + b] = s_logp;
}

// ---------------------------------------------------------------------------
extern "C" void kernel_launch(void* const* d_in, const int* in_sizes, int n_in,
                              void* d_out, int out_size, void* d_ws, size_t ws_size,
                              hipStream_t stream) {
  (void)in_sizes; (void)n_in; (void)out_size; (void)ws_size;
  const float* x         = (const float*)d_in[0];
  const int*   start_idx = (const int*)d_in[1];
  const float* a_emb_W   = (const float*)d_in[4];
  const float* a_emb_b   = (const float*)d_in[5];
  const float* a_attn_W  = (const float*)d_in[6];
  const float* a_attn_b  = (const float*)d_in[7];
  const float* a_ff_W1   = (const float*)d_in[8];
  const float* a_ff_b1   = (const float*)d_in[9];
  const float* a_ff_W2   = (const float*)d_in[10];
  const float* a_ff_b2   = (const float*)d_in[11];
  const float* a_ln      = (const float*)d_in[12];
  const float* s_emb_W   = (const float*)d_in[13];
  const float* s_emb_b   = (const float*)d_in[14];
  const float* s_attn_W  = (const float*)d_in[15];
  const float* s_attn_b  = (const float*)d_in[16];
  const float* s_ff_W1   = (const float*)d_in[17];
  const float* s_ff_b1   = (const float*)d_in[18];
  const float* s_ff_W2   = (const float*)d_in[19];
  const float* s_ff_b2   = (const float*)d_in[20];
  const float* s_ln      = (const float*)d_in[21];
  const float* WK        = (const float*)d_in[22];
  const float* bKp       = (const float*)d_in[23];
  const float* WV        = (const float*)d_in[24];
  const float* bVp       = (const float*)d_in[25];
  const float* Wq_mlp    = (const float*)d_in[26];
  const float* bq_mlp    = (const float*)d_in[27];
  const float* d_Wq      = (const float*)d_in[28];
  const float* d_bq      = (const float*)d_in[29];
  const float* d_Wo      = (const float*)d_in[30];
  const float* d_bo      = (const float*)d_in[31];

  _Float16* ws = (_Float16*)d_ws;
  _Float16* aW16 = ws;                 // packed (1024 x 128) = 131072
  _Float16* aF1  = ws + 131072;        // packed (256 x 512)  = 131072
  _Float16* aF2  = ws + 262144;        // packed (1024 x 128) = 131072
  _Float16* sW16 = ws + 393216;
  _Float16* sF1  = ws + 524288;
  _Float16* sF2  = ws + 655360;
  _Float16* WK16 = ws + 786432;        // packed (256 x 256) = 65536
  _Float16* WV16 = ws + 851968;

  const int NW = 131072;
  pack_bfrag<<<(NW + 255) / 256, 256, 0, stream>>>(a_attn_W, aW16, 1024, 128);
  pack_bfrag<<<(NW + 255) / 256, 256, 0, stream>>>(a_ff_W1,  aF1,   256, 512);
  pack_bfrag<<<(NW + 255) / 256, 256, 0, stream>>>(a_ff_W2,  aF2,  1024, 128);
  pack_bfrag<<<(NW + 255) / 256, 256, 0, stream>>>(s_attn_W, sW16, 1024, 128);
  pack_bfrag<<<(NW + 255) / 256, 256, 0, stream>>>(s_ff_W1,  sF1,   256, 512);
  pack_bfrag<<<(NW + 255) / 256, 256, 0, stream>>>(s_ff_W2,  sF2,  1024, 128);
  pack_bfrag<<<(65536 + 255) / 256, 256, 0, stream>>>(WK, WK16, 256, 256);
  pack_bfrag<<<(65536 + 255) / 256, 256, 0, stream>>>(WV, WV16, 256, 256);

  tsp_solver_kernel<<<NB, 256, 0, stream>>>(
      x, start_idx,
      a_emb_W, a_emb_b, aW16, a_attn_b, aF1, a_ff_b1, aF2, a_ff_b2, a_ln,
      s_emb_W, s_emb_b, sW16, s_attn_b, sF1, s_ff_b1, sF2, s_ff_b2, s_ln,
      WK16, bKp, WV16, bVp, Wq_mlp, bq_mlp,
      d_Wq, d_bq, d_Wo, d_bo,
      (float*)d_out);
}
